// SelfAttention_35072702939765
// MI455X (gfx1250) — matrix operand order repaired and checked by reading, not compiled, not run on hardware
//
#include <hip/hip_runtime.h>

// Problem constants (from reference): B=8, C=2304, H=W=48, N=H*W=2304 == C.
constexpr int Bb   = 8;
constexpr int Cdim = 2304;
constexpr int Ndim = 2304;

constexpr int TB = 256;           // 8 waves of 32
constexpr int BM = 128, BN = 128, BK = 32;
constexpr int NT = Cdim / BK;     // 72 K-tiles
constexpr int SA  = BK + 8;       // As row stride in f16 (80 B, 16B multiple)
constexpr int SBH = BN + 8;       // Bs row stride in f16 (272 B, 16B multiple)

typedef _Float16     v16h  __attribute__((ext_vector_type(16)));
typedef float        v8f   __attribute__((ext_vector_type(8)));
typedef float        v4f   __attribute__((ext_vector_type(4)));
typedef unsigned int v4u   __attribute__((ext_vector_type(4)));
typedef unsigned int v2u   __attribute__((ext_vector_type(2)));
typedef unsigned int u32x4 __attribute__((ext_vector_type(4)));
typedef int          i32x8 __attribute__((ext_vector_type(8)));
typedef int          i32x4 __attribute__((ext_vector_type(4)));

union FragH { v4u u4[2]; unsigned int u[8]; v16h h; };

// ---- CDNA5 Tensor Data Mover availability (this toolchain: 6-arg form) ----
#if defined(__has_builtin)
#if __has_builtin(__builtin_amdgcn_tensor_load_to_lds) && \
    __has_builtin(__builtin_amdgcn_s_wait_tensorcnt)
#define TDM_OK 1
#endif
#endif
#ifndef TDM_OK
#define TDM_OK 0
#endif

// Fallback per-lane async DMA: 16 bytes global -> LDS (ASYNCcnt tracked).
__device__ __forceinline__ void async_b128(const void* gp, const void* lp) {
  unsigned lds = (unsigned)(size_t)lp;   // LDS offset = low 32b of generic ptr
  unsigned long long ga = (unsigned long long)(size_t)gp;
  asm volatile("global_load_async_to_lds_b128 %0, %1, off"
               :: "v"(lds), "v"(ga) : "memory");
}

#if TDM_OK
// Issue one TDM 2D tile load: global (row-major, strided) -> LDS (padded rows).
// D# per CDNA5 ISA. All inputs workgroup-uniform.
__device__ __forceinline__ void tdm_load_2d(const void* gtile, unsigned lds_addr,
                                            unsigned g1w0,      // pad/data_size cfg
                                            unsigned tdim0, unsigned tdim1,
                                            unsigned tile0, unsigned tile1,
                                            unsigned stride0) {
  unsigned long long ga = (unsigned long long)(size_t)gtile;
  u32x4 g0;
  g0[0] = 1u;                                   // count=1, user descriptor
  g0[1] = lds_addr;                             // lds_addr[31:0]
  g0[2] = (unsigned)ga;                         // global_addr[31:0]
  g0[3] = (unsigned)(ga >> 32) | 0x80000000u;   // global_addr[56:32] | type=2
  i32x8 g1;
  g1[0] = (int)g1w0;                                        // mask/dsize/pad cfg
  g1[1] = (int)(tdim0 << 16);                               // tensor_dim0[15:0]
  g1[2] = (int)((tdim0 >> 16) | (tdim1 << 16));             // d0 hi | d1 lo
  g1[3] = (int)((tdim1 >> 16) | (tile0 << 16));             // d1 hi | tile_dim0
  g1[4] = (int)(tile1 & 0xFFFFu);                           // tile_dim1 (dim2=0)
  g1[5] = (int)stride0;                                     // dim0_stride lo
  g1[6] = 0;                                                // stride hi | d1str
  g1[7] = 0;
  i32x4 z4 = {0, 0, 0, 0};                      // 2D tensor: groups 2/3 unused
  i32x8 z8 = {0, 0, 0, 0, 0, 0, 0, 0};
  __builtin_amdgcn_tensor_load_to_lds(g0, g1, z4, z4, z8, 0);
}
// group1 word0: data_size=2B (bit16), pad_enable (bit20),
// pad_interval code (bits 24:22: 3 -> 16 dwords, 5 -> 64 dwords),
// pad_amount code (bits 31:25: 3 -> 4 dwords = 8 f16).
constexpr unsigned G1W0_A = (1u << 16) | (1u << 20) | (3u << 22) | (3u << 25);
constexpr unsigned G1W0_B = (1u << 16) | (1u << 20) | (5u << 22) | (3u << 25);
#endif

__device__ __forceinline__ void wait_tiles() {
#if TDM_OK
  __builtin_amdgcn_s_wait_tensorcnt(0);
#else
  asm volatile("s_wait_asynccnt 0x0" ::: "memory");
#endif
}

// ---------------------------------------------------------------------------
// f32 -> f16 convert (vector loads, packed stores)
// ---------------------------------------------------------------------------
__global__ __launch_bounds__(256) void cvt_f32_f16(const float* __restrict__ s,
                                                   _Float16* __restrict__ d,
                                                   size_t n) {
  size_t i    = ((size_t)blockIdx.x * blockDim.x + threadIdx.x) * 4;
  size_t step = (size_t)gridDim.x * blockDim.x * 4;
  for (; i < n; i += step) {
    v4f v = *(const v4f*)(s + i);
    union { _Float16 h[2]; unsigned int u; } p0, p1;
    p0.h[0] = (_Float16)v[0]; p0.h[1] = (_Float16)v[1];
    p1.h[0] = (_Float16)v[2]; p1.h[1] = (_Float16)v[3];
    v2u o; o[0] = p0.u; o[1] = p1.u;
    *(v2u*)(d + i) = o;
  }
}

// ---------------------------------------------------------------------------
// Tiled f16 WMMA GEMM: D[2304,2304] = A(row-major) * B(row-major), fp32 acc.
// Double-buffered LDS; tiles staged by the Tensor Data Mover (wave 0 issues
// one TENSOR_LOAD_TO_LDS per tile; hardware applies the LDS row padding), or
// by per-lane GLOBAL_LOAD_ASYNC_TO_LDS_B128 when the TDM builtin is absent.
// MODE 0: +bias[row], store f16        (Q/K/V projections)
// MODE 1: store f32                    (energy)
// MODE 2: gamma*acc + resid, store f32 (attn*V + residual -> d_out)
// ---------------------------------------------------------------------------
template <int MODE>
__global__ __launch_bounds__(TB) void gemm_wmma(
    const _Float16* __restrict__ Aall, size_t strideA,
    const _Float16* __restrict__ Ball, size_t strideB,
    void* __restrict__ Optr, size_t strideO,
    const float* __restrict__ aux,      // MODE0: bias[C]; MODE2: residual x (f32)
    const float* __restrict__ gammaPtr) // MODE2 only
{
  __shared__ __align__(16) _Float16 As[2][BM * SA];
  __shared__ __align__(16) _Float16 Bs[2][BK * SBH];

  const int tid  = threadIdx.x;
  const int lane = tid & 31;
  const int wv   = tid >> 5;
  const int wm   = wv >> 2;    // 0..1 : 64-row slab
  const int wn   = wv & 3;     // 0..3 : 32-col slab
  const int l16  = lane & 15;
  const int lhi  = lane >> 4;  // 0 or 1

  const int m0 = blockIdx.y * BM;
  const int n0 = blockIdx.x * BN;
  const int b  = blockIdx.z;

  const _Float16* gA = Aall + (size_t)b * strideA;
  const _Float16* gB = Ball + (size_t)b * strideB;

  v8f acc[4][2] = {};

  // --- stage one K-tile pair (A 128x32, B 32x128) into LDS buffer `buf` ---
  auto stage = [&](int kt, int buf) {
    const int k0 = kt * BK;
#if TDM_OK
    if (wv == 0) {
      tdm_load_2d(gA + (size_t)m0 * Cdim + k0, (unsigned)(size_t)&As[buf][0],
                  G1W0_A, (unsigned)Cdim, (unsigned)Cdim, BK, BM, (unsigned)Cdim);
      tdm_load_2d(gB + (size_t)k0 * Ndim + n0, (unsigned)(size_t)&Bs[buf][0],
                  G1W0_B, (unsigned)Ndim, (unsigned)Cdim, BN, BK, (unsigned)Ndim);
    }
#else
#pragma unroll
    for (int i = 0; i < 2; ++i) {
      const int c   = tid + i * TB;     // 0..511
      const int ar  = c >> 2;           // A row 0..127
      const int ac8 = (c & 3) * 8;      // A col chunk
      async_b128(gA + (size_t)(m0 + ar) * Cdim + (k0 + ac8),
                 &As[buf][ar * SA + ac8]);
      const int br  = c >> 4;           // B row 0..31
      const int bc8 = (c & 15) * 8;     // B col chunk
      async_b128(gB + (size_t)(k0 + br) * Ndim + (n0 + bc8),
                 &Bs[buf][br * SBH + bc8]);
    }
#endif
  };

  stage(0, 0);
  wait_tiles();
  __syncthreads();

  const unsigned bsel = 0x05040100u;  // v_perm selector base (even column)

  for (int kt = 0; kt < NT; ++kt) {
    const int cur = kt & 1;
    if (kt + 1 < NT) stage(kt + 1, cur ^ 1);   // DMA next tile during compute

    // ---- fragments: contiguous k order, k = 16*lhi + s, for A and B ----
    FragH fa[4], fb[2];
    const int kst = lhi * 16;  // A: lanes 0-15 -> K{0..15}; 16-31 -> K{16..31}
#pragma unroll
    for (int i = 0; i < 4; ++i) {
      const int row = wm * 64 + i * 16 + l16;
      fa[i].u4[0] = *(const v4u*)(&As[cur][row * SA + kst]);
      fa[i].u4[1] = *(const v4u*)(&As[cur][row * SA + kst + 8]);
    }
#pragma unroll
    for (int j = 0; j < 2; ++j) {
      const int col  = wn * 32 + j * 16 + l16;
      const int cole = col & ~1;                       // even column pair base
      const unsigned sel = bsel + (unsigned)(col & 1) * 0x02020202u;
      const int kh = lhi * 16;                         // B: lanes 16-31 -> K+16
#pragma unroll
      for (int v = 0; v < 8; ++v) {
        unsigned r0 = *(const unsigned*)(&Bs[cur][(kh + 2 * v)     * SBH + cole]);
        unsigned r1 = *(const unsigned*)(&Bs[cur][(kh + 2 * v + 1) * SBH + cole]);
        fb[j].u[v] = __builtin_amdgcn_perm(r1, r0, sel);  // {r0.h, r1.h}
      }
    }

    // ---- 8 WMMAs per K-step ----
#pragma unroll
    for (int i = 0; i < 4; ++i)
#pragma unroll
      for (int j = 0; j < 2; ++j)
        acc[i][j] = __builtin_amdgcn_wmma_f32_16x16x32_f16(
            false, fa[i].h, false, fb[j].h, (short)0, acc[i][j], false, false);

    wait_tiles();       // next tile fully landed in LDS (issuing wave)
    __syncthreads();    // publish; all waves done reading current buffers
  }

  // ---- epilogue: C/D layout -> lane col = n0+l16, VGPRr row = r + 8*lhi ----
  float g = 0.f;
  if (MODE == 2) g = gammaPtr[0];
#pragma unroll
  for (int i = 0; i < 4; ++i) {
#pragma unroll
    for (int j = 0; j < 2; ++j) {
      const int col = n0 + wn * 32 + j * 16 + l16;
#pragma unroll
      for (int r = 0; r < 8; ++r) {
        const int row = m0 + wm * 64 + i * 16 + r + lhi * 8;
        float v = acc[i][j][r];
        if (MODE == 0) {
          v += aux[row];
          ((_Float16*)Optr)[(size_t)b * strideO + (size_t)row * Ndim + col] = (_Float16)v;
        } else if (MODE == 1) {
          ((float*)Optr)[(size_t)b * strideO + (size_t)row * Ndim + col] = v;
        } else {
          const size_t idx = (size_t)b * strideO + (size_t)row * Ndim + col;
          ((float*)Optr)[idx] = g * v + aux[idx];
        }
      }
    }
  }
}

// ---------------------------------------------------------------------------
// Row softmax: one 256-thread block per row of 2304 f32 -> f16 probabilities
// ---------------------------------------------------------------------------
__global__ __launch_bounds__(256) void softmax_rows(const float* __restrict__ E,
                                                    _Float16* __restrict__ A,
                                                    int n) {
  __shared__ float red[8];
  const int tid = threadIdx.x, lane = tid & 31, wv = tid >> 5;
  const size_t base = (size_t)blockIdx.x * n;

  float m = -3.402823e38f;
  for (int j = tid; j < n; j += 256) m = fmaxf(m, E[base + j]);
#pragma unroll
  for (int o = 16; o > 0; o >>= 1) m = fmaxf(m, __shfl_xor(m, o, 32));
  if (lane == 0) red[wv] = m;
  __syncthreads();
  float mx = red[0];
#pragma unroll
  for (int i = 1; i < 8; ++i) mx = fmaxf(mx, red[i]);
  __syncthreads();

  float s = 0.f;
  for (int j = tid; j < n; j += 256) s += __expf(E[base + j] - mx);
#pragma unroll
  for (int o = 16; o > 0; o >>= 1) s += __shfl_xor(s, o, 32);
  if (lane == 0) red[wv] = s;
  __syncthreads();
  float tot = 0.f;
#pragma unroll
  for (int i = 0; i < 8; ++i) tot += red[i];
  const float inv = 1.0f / tot;

  for (int j = tid; j < n; j += 256)
    A[base + j] = (_Float16)(__expf(E[base + j] - mx) * inv);
}

// ---------------------------------------------------------------------------
extern "C" void kernel_launch(void* const* d_in, const int* in_sizes, int n_in,
                              void* d_out, int out_size, void* d_ws, size_t ws_size,
                              hipStream_t stream) {
  (void)in_sizes; (void)n_in; (void)out_size; (void)ws_size;

  const float* x  = (const float*)d_in[0];
  const float* Wq = (const float*)d_in[1];
  const float* bq = (const float*)d_in[2];
  const float* Wk = (const float*)d_in[3];
  const float* bk = (const float*)d_in[4];
  const float* Wv = (const float*)d_in[5];
  const float* bv = (const float*)d_in[6];
  const float* gm = (const float*)d_in[7];

  const size_t CN  = (size_t)Cdim * Ndim;  // per-batch matrix elements
  const size_t BCN = (size_t)Bb * CN;

  char* p = (char*)d_ws;
  _Float16* xb  = (_Float16*)p; p += BCN * 2;
  _Float16* wqb = (_Float16*)p; p += CN * 2;
  _Float16* wkb = (_Float16*)p; p += CN * 2;
  _Float16* wvb = (_Float16*)p; p += CN * 2;
  _Float16* qb  = (_Float16*)p; p += BCN * 2;
  _Float16* kb  = (_Float16*)p; p += BCN * 2;
  _Float16* vb  = (_Float16*)p; p += BCN * 2;
  float*    E   = (float*)p;    p += BCN * 4;
  _Float16* attnb = xb;  // xb is dead after the Q/K/V GEMMs -> reuse for attn

  // f32 -> f16 staging
  cvt_f32_f16<<<dim3(4096), dim3(256), 0, stream>>>(x,  xb,  BCN);
  cvt_f32_f16<<<dim3(512),  dim3(256), 0, stream>>>(Wq, wqb, CN);
  cvt_f32_f16<<<dim3(512),  dim3(256), 0, stream>>>(Wk, wkb, CN);
  cvt_f32_f16<<<dim3(512),  dim3(256), 0, stream>>>(Wv, wvb, CN);

  const dim3 gg(Ndim / BN, Cdim / BM, Bb);  // 18 x 18 x 8

  // Q/K/V projections: W (shared across batch) x X + bias -> f16
  gemm_wmma<0><<<gg, TB, 0, stream>>>(wqb, 0, xb, CN, qb, CN, bq, nullptr);
  gemm_wmma<0><<<gg, TB, 0, stream>>>(wkb, 0, xb, CN, kb, CN, bk, nullptr);
  gemm_wmma<0><<<gg, TB, 0, stream>>>(wvb, 0, xb, CN, vb, CN, bv, nullptr);

  // energy = Q x K (f32)
  gemm_wmma<1><<<gg, TB, 0, stream>>>(qb, CN, kb, CN, E, CN, nullptr, nullptr);

  // softmax rows -> f16 attn
  softmax_rows<<<dim3(Bb * Cdim), dim3(256), 0, stream>>>(E, attnb, Ndim);

  // out = gamma * (attn x V) + x  (f32 -> d_out)
  gemm_wmma<2><<<gg, TB, 0, stream>>>(attnb, CN, vb, CN, d_out, CN, x, gm);
}
